// GoSpecificWattiPooling_85280870629541
// MI455X (gfx1250) — compile-verified, matrix operand order checked by reading, not run on hardware
//
#include <hip/hip_runtime.h>
#include <hip/hip_bf16.h>

// MI455X (gfx1250) wave32 + WMMA bf16 GO-token attention pooling.
//
// Pipeline:
//   transpose_h_bf16: Ht = bf16(H)^T        [B,1024,2048]  (ws, 32 MB)  (d-major)
//   proj_gemm_bf16:   Qb = bf16(G @ Wq^T)   [B,1024,256]   (ws,  4 MB)
//                     Kb = bf16(H @ Wk^T)   [B,2048,256]   (ws,  8 MB)
//   attn_kernel:      flash attention, O = softmax(Q K^T / 16) @ H
//
// Matmuls: v_wmma_f32_16x16x32_bf16 (fp32 accumulate). Both K and H chunk
// staging use global_load_async_to_lds_b128 (ASYNCcnt) — pure async copies,
// no VALU/DS in the staging path.

typedef __attribute__((ext_vector_type(16))) __bf16 v16bf;
typedef __attribute__((ext_vector_type(8)))  float  v8f;

union BF16x16 {
    v16bf v;
    unsigned int u[8];
};

// fp32 -> bf16 single value, round-to-nearest-even
static __device__ __forceinline__ unsigned short f2bf(float f) {
    unsigned u = __float_as_uint(f);
    u += 0x7FFFu + ((u >> 16) & 1u);
    return (unsigned short)(u >> 16);
}

// fp32 pair -> packed bf16x2 (lowers to v_cvt_pk_bf16_f32)
static __device__ __forceinline__ unsigned pack2(float x, float y) {
    union { __hip_bfloat162 b; unsigned u; } c;
    c.b = __float22bfloat162_rn(make_float2(x, y));
    return c.u;
}

// K-index inside a 16x32 bf16 A/B fragment for VGPR v, lane-half h.
// (ISA 7.12.2) Pairs (e=0,1) are consecutive K values -> one aligned 32-bit load.
static __device__ __forceinline__ int kidx(int v, int h) {
    return ((v >> 2) << 4) + ((v & 3) << 1) + (h << 3);
}

// ---------------------------------------------------------------------------
// Kernel 0: Ht[b][d][l] = bf16(H[b][l][d]).  LDS-tiled 32x32 transpose.
// ---------------------------------------------------------------------------
__global__ __launch_bounds__(256) void transpose_h_bf16(
    const float* __restrict__ H, unsigned short* __restrict__ Ht)
{
    __shared__ unsigned short tile[32 * 33];   // padded to dodge bank conflicts

    const int l0 = blockIdx.x * 32;   // 2048 / 32 = 64
    const int d0 = blockIdx.y * 32;   // 1024 / 32 = 32
    const int b  = blockIdx.z;
    const int tid = threadIdx.x;

    const float* Hb = H + ((size_t)b * 2048) * 1024;
    unsigned short* Tb = Ht + ((size_t)b * 1024) * 2048;

    // Read 32x32 fp32 tile, convert, store transposed into LDS.
#pragma unroll
    for (int i = 0; i < 2; ++i) {
        const int idx = tid + i * 256;       // 0..511 float2 groups
        const int d2  = idx & 15;            // pair along d
        const int l   = idx >> 4;            // 0..31
        const float2 v = *(const float2*)(Hb + (size_t)(l0 + l) * 1024 + d0 + 2 * d2);
        tile[(2 * d2 + 0) * 33 + l] = f2bf(v.x);
        tile[(2 * d2 + 1) * 33 + l] = f2bf(v.y);
    }
    __syncthreads();

    // Write out rows of Ht (contiguous along l), packed u32 pairs.
#pragma unroll
    for (int i = 0; i < 2; ++i) {
        const int idx = tid + i * 256;       // 0..511 u32 outputs
        const int lp  = idx & 15;            // l-pair
        const int d   = idx >> 4;            // 0..31
        const unsigned lo = tile[d * 33 + 2 * lp];
        const unsigned hi = tile[d * 33 + 2 * lp + 1];
        *(unsigned*)(Tb + (size_t)(d0 + d) * 2048 + l0 + 2 * lp) = lo | (hi << 16);
    }
}

// ---------------------------------------------------------------------------
// Kernel 1: C_bf16[b, m, n] = sum_k A[b, m, k] * W[n, k]    (N fixed = 256)
// One wave (32 threads) per 16x16 output tile.
// ---------------------------------------------------------------------------
__global__ __launch_bounds__(32) void proj_gemm_bf16(
    const float* __restrict__ A, const float* __restrict__ W,
    unsigned short* __restrict__ C, int M, int KK)
{
    const int n0   = blockIdx.x * 16;
    const int m0   = blockIdx.y * 16;
    const int b    = blockIdx.z;
    const int lane = threadIdx.x & 31;
    const int h    = lane >> 4;    // lane half
    const int lm   = lane & 15;    // M (A) / N (B) index for this lane

    const float* Ab = A + (size_t)b * M * KK;
    unsigned short* Cb = C + (size_t)b * M * 256;

    v8f acc = {};
    for (int ks = 0; ks < KK; ks += 32) {
        BF16x16 af, bf;
#pragma unroll
        for (int v = 0; v < 8; ++v) {
            const int k = ks + kidx(v, h);
            const float2 av = *(const float2*)(Ab + (size_t)(m0 + lm) * KK + k);
            const float2 wv = *(const float2*)(W  + (size_t)(n0 + lm) * KK + k);
            af.u[v] = pack2(av.x, av.y);
            bf.u[v] = pack2(wv.x, wv.y);
        }
        acc = __builtin_amdgcn_wmma_f32_16x16x32_bf16(
            false, af.v, false, bf.v, (short)0, acc, false, false);
    }
    // C/D layout: VGPR r -> M = r + 8*h, N = lane&15
#pragma unroll
    for (int r = 0; r < 8; ++r) {
        const int mo = m0 + r + 8 * h;
        Cb[(size_t)mo * 256 + n0 + lm] = f2bf(acc[r]);
    }
}

// ---------------------------------------------------------------------------
// Kernel 2: flash attention. Block = 16 T-rows x 512 Dh-cols, 8 waves.
// Waves 0,1 compute the 16x32 logit tile; all 8 waves own 4 output N-tiles.
// ---------------------------------------------------------------------------
#define LC 32  // L-chunk (= WMMA K for the P*H matmul)

__global__ __launch_bounds__(256) void attn_kernel(
    const unsigned short* __restrict__ Htg,   // [B,1024,2048] bf16, d-major
    const unsigned short* __restrict__ Qb,
    const unsigned short* __restrict__ Kb,
    float* __restrict__ Out)
{
    __shared__ unsigned short Kc[LC * 256];   // K chunk, row-major [l][p]  (16 KB)
    __shared__ unsigned short Hc[512 * LC];   // H chunk, [d_local][l]      (32 KB)
    __shared__ float          Sbuf[16 * LC];  // logits                     (2 KB)
    __shared__ unsigned short Pbuf[16 * LC];  // exp(S - m), bf16           (1 KB)
    __shared__ float mrow[16], lrow[16], srow[16];

    const int dhalf = blockIdx.x;          // which 512-wide half of Dh
    const int t0    = blockIdx.y * 16;
    const int b     = blockIdx.z;
    const int tid   = threadIdx.x;
    const int lane  = tid & 31;
    const int wave  = tid >> 5;
    const int h     = lane >> 4;
    const int lm    = lane & 15;
    const int dbase = dhalf * 512;
    const float scale = 0.0625f;           // 256^-0.5

    const unsigned short* Kbase = Kb  + (size_t)b * 2048 * 256;
    const unsigned short* Hbase = Htg + ((size_t)b * 1024 + dbase) * 2048;

    // Q fragments: 8 K-slices of the 16x256 Q tile (only waves 0,1 use them)
    BF16x16 qf[8];
    if (wave < 2) {
        const unsigned short* Qrow = Qb + ((size_t)b * 1024 + t0) * 256;
#pragma unroll
        for (int s = 0; s < 8; ++s)
#pragma unroll
            for (int v = 0; v < 8; ++v)
                qf[s].u[v] = *(const unsigned int*)(Qrow + (size_t)lm * 256 + s * 32 + kidx(v, h));
    }

    if (tid < 16) { mrow[tid] = -1e30f; lrow[tid] = 0.0f; }

    v8f acc[4] = {};

    for (int l0 = 0; l0 < 2048; l0 += LC) {
        // --- stage K chunk: 16 KB contiguous, async global->LDS b128 ---
        {
            const unsigned short* src = Kbase + (size_t)l0 * 256;
#pragma unroll
            for (int i = 0; i < 4; ++i) {
                const int idx = (tid + i * 256) * 8;  // 8 bf16 per b128 transfer
                const unsigned lds_addr = (unsigned)(size_t)&Kc[idx];
                const unsigned long long gaddr = (unsigned long long)(size_t)(src + idx);
                asm volatile("global_load_async_to_lds_b128 %0, %1, off"
                             :: "v"(lds_addr), "v"(gaddr) : "memory");
            }
        }
        // --- stage H chunk: Htg rows are contiguous along l -> async b128 ---
        {
#pragma unroll
            for (int i = 0; i < 8; ++i) {
                const int idx = tid + i * 256;   // 0..2047 transfers of 16 B
                const int dl  = idx >> 2;        // local d row (0..511)
                const int g   = idx & 3;         // 16B group within the 32-l row
                const unsigned lds_addr = (unsigned)(size_t)&Hc[dl * LC + 8 * g];
                const unsigned long long gaddr =
                    (unsigned long long)(size_t)(Hbase + (size_t)dl * 2048 + l0 + 8 * g);
                asm volatile("global_load_async_to_lds_b128 %0, %1, off"
                             :: "v"(lds_addr), "v"(gaddr) : "memory");
            }
        }
        // Each wave drains its own async transfers (ASYNCcnt is per-wave),
        // then the barrier publishes all waves' LDS writes.
        asm volatile("s_wait_asynccnt 0" ::: "memory");
        __syncthreads();

        // --- S = Q K^T * scale (waves 0,1; wave w owns columns w*16..w*16+15) ---
        if (wave < 2) {
            v8f s = {};
#pragma unroll
            for (int ks = 0; ks < 8; ++ks) {
                BF16x16 bf;
                const int j = wave * 16 + lm;   // L-row within chunk
#pragma unroll
                for (int v = 0; v < 8; ++v)
                    bf.u[v] = *(const unsigned int*)&Kc[j * 256 + ks * 32 + kidx(v, h)];
                s = __builtin_amdgcn_wmma_f32_16x16x32_bf16(
                    false, qf[ks].v, false, bf.v, (short)0, s, false, false);
            }
#pragma unroll
            for (int r = 0; r < 8; ++r)
                Sbuf[(r + 8 * h) * LC + wave * 16 + lm] = s[r] * scale;
        }
        __syncthreads();

        // --- online softmax stats, one row per thread ---
        if (tid < 16) {
            const float mold = mrow[tid];
            float cmax = -1e30f;
#pragma unroll
            for (int j = 0; j < LC; ++j) cmax = fmaxf(cmax, Sbuf[tid * LC + j]);
            const float mnew = fmaxf(mold, cmax);
            const float scl  = __expf(mold - mnew);
            float sum = 0.0f;
#pragma unroll
            for (int j = 0; j < LC; ++j) {
                const float p = __expf(Sbuf[tid * LC + j] - mnew);
                Pbuf[tid * LC + j] = f2bf(p);
                sum += p;
            }
            lrow[tid] = lrow[tid] * scl + sum;
            mrow[tid] = mnew;
            srow[tid] = scl;
        }
        __syncthreads();

        // --- rescale accumulators, then O += P * H_chunk (all 8 waves) ---
        float sc[8];
#pragma unroll
        for (int r = 0; r < 8; ++r) sc[r] = srow[r + 8 * h];
#pragma unroll
        for (int nt = 0; nt < 4; ++nt)
#pragma unroll
            for (int r = 0; r < 8; ++r) acc[nt][r] *= sc[r];

        BF16x16 pf;
#pragma unroll
        for (int v = 0; v < 8; ++v)
            pf.u[v] = *(const unsigned int*)&Pbuf[lm * LC + kidx(v, h)];

#pragma unroll
        for (int nt = 0; nt < 4; ++nt) {
            BF16x16 bf;
            const int dloc = wave * 64 + nt * 16 + lm;   // local Dh column
#pragma unroll
            for (int v = 0; v < 8; ++v)
                bf.u[v] = *(const unsigned int*)&Hc[dloc * LC + kidx(v, h)];
            acc[nt] = __builtin_amdgcn_wmma_f32_16x16x32_bf16(
                false, pf.v, false, bf.v, (short)0, acc[nt], false, false);
        }
        __syncthreads();
    }

    // --- epilogue: divide by clamped denominator, store fp32 ---
    float inv[8];
#pragma unroll
    for (int r = 0; r < 8; ++r)
        inv[r] = 1.0f / fmaxf(lrow[r + 8 * h], 1e-20f);

    float* Ob = Out + ((size_t)b * 1024 + t0) * 1024;
#pragma unroll
    for (int nt = 0; nt < 4; ++nt) {
        const int dcol = dbase + wave * 64 + nt * 16 + lm;
#pragma unroll
        for (int r = 0; r < 8; ++r)
            Ob[(size_t)(r + 8 * h) * 1024 + dcol] = acc[nt][r] * inv[r];
    }
}

extern "C" void kernel_launch(void* const* d_in, const int* in_sizes, int n_in,
                              void* d_out, int out_size, void* d_ws, size_t ws_size,
                              hipStream_t stream)
{
    const float* Hp = (const float*)d_in[0];   // [8,2048,1024]
    const float* Gp = (const float*)d_in[1];   // [8,1024,768]
    const float* Wq = (const float*)d_in[2];   // [256,768]
    const float* Wk = (const float*)d_in[3];   // [256,1024]
    float* Out = (float*)d_out;                // [8,1024,1024]

    unsigned short* Qb = (unsigned short*)d_ws;              // 8*1024*256 bf16 ( 4 MB)
    unsigned short* Kb = Qb + (size_t)8 * 1024 * 256;        // 8*2048*256 bf16 ( 8 MB)
    unsigned short* Ht = Kb + (size_t)8 * 2048 * 256;        // 8*1024*2048 bf16 (32 MB)

    // Ht = bf16(H)^T  (d-major rows, contiguous along l)
    transpose_h_bf16<<<dim3(64, 32, 8), 256, 0, stream>>>(Hp, Ht);
    // Q = G @ Wq^T  (M=1024, K=768)
    proj_gemm_bf16<<<dim3(16, 64, 8), 32, 0, stream>>>(Gp, Wq, Qb, 1024, 768);
    // K = H @ Wk^T  (M=2048, K=1024)
    proj_gemm_bf16<<<dim3(16, 128, 8), 32, 0, stream>>>(Hp, Wk, Kb, 2048, 1024);
    // Flash attention over L, Dh split across blockIdx.x
    attn_kernel<<<dim3(2, 64, 8), 256, 0, stream>>>(Ht, Qb, Kb, Out);
}
